// RegionProposalNetwork_21242908246058
// MI455X (gfx1250) — matrix operand order repaired and checked by reading, not compiled, not run on hardware
//
#include <hip/hip_runtime.h>
#include <hip/hip_bf16.h>
#include <stdint.h>

// ---------------------------------------------------------------------------
// Problem constants (from reference)
// ---------------------------------------------------------------------------
#define BATCH   4
#define CIN     2048
#define HH      32
#define WW      32
#define HID     512
#define NA      9
#define NG      20
#define MDIM    (BATCH * HH * WW)      // 4096  (GEMM M)
#define KFULL   (CIN * 9)              // 18432 (GEMM K), ordered k = tap*2048 + c
#define NANCH   (HH * WW * NA)         // 9216 anchors per image
#define SCALE_F 32.0f
#define POS_T   0.7f
#define NEG_T   0.3f
#define W_CONF  1.0f
#define W_REG   5.0f
#define EPSF    1e-8f

typedef unsigned short u16;
typedef __attribute__((ext_vector_type(16))) __bf16 v16bf;
typedef __attribute__((ext_vector_type(8)))  float  v8f;

struct alignas(16) U4 { unsigned x, y, z, w; };

__device__ __forceinline__ u16 f2bf(float f) {
    unsigned u = __float_as_uint(f);
    unsigned r = u + 0x7FFFu + ((u >> 16) & 1u);   // round-to-nearest-even
    return (u16)(r >> 16);
}
__device__ __forceinline__ float bf2f(u16 h) {
    return __uint_as_float(((unsigned)h) << 16);
}

// ---------------------------------------------------------------------------
// Pack conv1 weights [HID][CIN][3][3] fp32 -> Wpack[oc][tap*2048+c] bf16
// ---------------------------------------------------------------------------
__global__ __launch_bounds__(256) void pack_w_kernel(const float* __restrict__ w1,
                                                     u16* __restrict__ Wpack) {
    int idx = blockIdx.x * 256 + threadIdx.x;
    if (idx >= HID * KFULL) return;
    int oc  = idx / KFULL;
    int k   = idx - oc * KFULL;
    int tap = k >> 11;          // k / 2048
    int c   = k & 2047;
    float v = w1[(size_t)oc * KFULL + (size_t)c * 9 + tap];
    Wpack[idx] = f2bf(v);
}

// ---------------------------------------------------------------------------
// Pack feature map [B][C][H][W] fp32 -> Apack[(b*1024 + y*32 + x)][c] bf16
// ---------------------------------------------------------------------------
__global__ __launch_bounds__(256) void pack_fm_kernel(const float* __restrict__ fm,
                                                      u16* __restrict__ Apack) {
    int idx = blockIdx.x * 256 + threadIdx.x;
    if (idx >= BATCH * CIN * HH * WW) return;
    int b = idx >> 21;              // / (2048*1024)
    int c = (idx >> 10) & 2047;
    int p = idx & 1023;             // y*32 + x
    Apack[((size_t)(b << 10) + p) * CIN + c] = f2bf(fm[idx]);
}

// ---------------------------------------------------------------------------
// Implicit-GEMM 3x3 conv with bf16 WMMA, fp32 accumulate, fused bias+ReLU.
//   C[m][oc] = relu( sum_k A[m][k] * W[oc][k] + bias[oc] )
// Block tile 128(M) x 128(N), 8 waves; wave tile 64x32 = 4x2 WMMA 16x16x32.
// Staging uses CDNA5 async global->LDS (ASYNCcnt) with LDS double buffering:
// one barrier per K-step; next tile streams into the other buffer while the
// current one feeds the WMMAs.
// ---------------------------------------------------------------------------
#define MT 128
#define NT 128
#define KT 32
#define KITERS (KFULL / KT)   // 576

__global__ __launch_bounds__(256) void conv_gemm_kernel(const u16* __restrict__ Apack,
                                                        const u16* __restrict__ Wpack,
                                                        const float* __restrict__ bias,
                                                        const u16* __restrict__ zpad,
                                                        u16* __restrict__ hid) {
    __shared__ __align__(16) u16 Alds[2][MT * KT];   // 2 x 8 KB
    __shared__ __align__(16) u16 Blds[2][NT * KT];   // 2 x 8 KB

    const int mtile = blockIdx.x;   // 0..31
    const int ntile = blockIdx.y;   // 0..3
    const int tid   = threadIdx.x;
    const int lane  = tid & 31;
    const int wid   = tid >> 5;     // 0..7
    const int wm    = wid & 1;      // 2 M-waves
    const int wn    = wid >> 1;     // 4 N-waves
    const int lrow  = lane & 15;
    const int lquad = lane >> 4;    // 0/1

    // staging coordinates (each thread stages 16 bf16 of A and 16 bf16 of B)
    const int row  = tid >> 1;      // 0..127
    const int half = tid & 1;       // 0/1 -> k subrange of 16
    const int mA   = mtile * MT + row;
    const int yA   = (mA & 1023) >> 5;
    const int xA   = mA & 31;
    const int nB   = ntile * NT + row;
    // Wpack row is k-linear: tap*2048 + c0 == kt*32
    const u16* wrow = Wpack + (size_t)nB * KFULL + (half << 4);

    // LDS byte offsets for the async engine (low 32 bits of generic LDS addr)
    const unsigned stoff  = (unsigned)((row * KT + (half << 4)) * sizeof(u16));
    const unsigned aBase0 = (unsigned)(size_t)&Alds[0][0];
    const unsigned aBase1 = (unsigned)(size_t)&Alds[1][0];
    const unsigned bBase0 = (unsigned)(size_t)&Blds[0][0];
    const unsigned bBase1 = (unsigned)(size_t)&Blds[1][0];

    auto issue = [&](int kt, int buf) {
        const int tap = kt >> 6;                   // filter tap 0..8
        const int c0  = (kt & 63) << 5;            // channel base
        const int t3  = (tap * 11) >> 5;           // tap / 3  (tap in 0..8)
        const int tdx = tap - 3 * t3;              // tap % 3
        const int yy  = yA + t3 - 1;
        const int xx  = xA + tdx - 1;
        const bool inb = ((unsigned)yy < (unsigned)HH) && ((unsigned)xx < (unsigned)WW);
        const u16* asrc = inb
            ? (Apack + ((size_t)(mA + (t3 - 1) * WW + (tdx - 1))) * CIN + c0 + (half << 4))
            : zpad;                                 // zero-filled 64B pad
        const u16* bsrc = wrow + (size_t)kt * KT;
        const unsigned aoff = (buf ? aBase1 : aBase0) + stoff;
        const unsigned boff = (buf ? bBase1 : bBase0) + stoff;
        unsigned long long ag = (unsigned long long)(size_t)asrc;
        unsigned long long bg = (unsigned long long)(size_t)bsrc;
        // INST_OFFSET applies to both the LDS and global address -> one base
        // covers both 16B halves of this thread's 32B stripe.
        asm volatile("global_load_async_to_lds_b128 %0, %1, off\n\t"
                     "global_load_async_to_lds_b128 %0, %1, off offset:16"
                     :: "v"(aoff), "v"(ag) : "memory");
        asm volatile("global_load_async_to_lds_b128 %0, %1, off\n\t"
                     "global_load_async_to_lds_b128 %0, %1, off offset:16"
                     :: "v"(boff), "v"(bg) : "memory");
    };

    v8f acc[4][2];
    #pragma unroll
    for (int i = 0; i < 4; ++i)
        #pragma unroll
        for (int j = 0; j < 2; ++j)
            #pragma unroll
            for (int e = 0; e < 8; ++e) acc[i][j][e] = 0.0f;

    issue(0, 0);

    for (int kt = 0; kt < KITERS; ++kt) {
        const int buf = kt & 1;
        // my async stores into LDS for tile kt have landed
        asm volatile("s_wait_asynccnt 0x0" ::: "memory");
        // everyone's tile kt is in LDS; everyone finished reading buf^1
        __syncthreads();
        if (kt + 1 < KITERS) issue(kt + 1, buf ^ 1);
        if (kt + 2 < KITERS)                       // L2 warm hint for tile kt+2
            __builtin_prefetch(wrow + (size_t)(kt + 2) * KT, 0, 1);

        const u16* Acur = &Alds[buf][0];
        const u16* Bcur = &Blds[buf][0];

        // A fragments: lane holds row (lrow), K split 8/8 by lane quadrant
        v16bf af[4];
        #pragma unroll
        for (int i = 0; i < 4; ++i) {
            const int r = wm * 64 + i * 16 + lrow;
            const u16* p = Acur + r * KT + lquad * 8;
            union { U4 u[2]; v16bf v; } t;
            t.u[0] = *(const U4*)(p);
            t.u[1] = *(const U4*)(p + 16);        // K+16 half
            af[i] = t.v;
        }
        // B fragments: lane holds col (lrow), K split 8/8 by lane quadrant
        v16bf bfr[2];
        #pragma unroll
        for (int j = 0; j < 2; ++j) {
            const int r = wn * 32 + j * 16 + lrow;
            const u16* p = Bcur + r * KT + lquad * 8;
            union { U4 u[2]; v16bf v; } t;
            t.u[0] = *(const U4*)(p);
            t.u[1] = *(const U4*)(p + 16);
            bfr[j] = t.v;
        }
        #pragma unroll
        for (int i = 0; i < 4; ++i)
            #pragma unroll
            for (int j = 0; j < 2; ++j)
                acc[i][j] = __builtin_amdgcn_wmma_f32_16x16x32_bf16(
                    false, af[i], false, bfr[j], (short)0, acc[i][j], false, false);
    }

    // Epilogue: bias + ReLU, store bf16 hid[m][oc]
    #pragma unroll
    for (int j = 0; j < 2; ++j) {
        const int ncol = ntile * NT + wn * 32 + j * 16 + lrow;
        const float bv = bias[ncol];
        #pragma unroll
        for (int i = 0; i < 4; ++i) {
            const int mbase = mtile * MT + wm * 64 + i * 16 + lquad * 8;
            #pragma unroll
            for (int v = 0; v < 8; ++v) {
                float val = acc[i][j][v] + bv;
                val = fmaxf(val, 0.0f);
                hid[(size_t)(mbase + v) * HID + ncol] = f2bf(val);
            }
        }
    }
}

// ---------------------------------------------------------------------------
// 1x1 heads: conf (9 ch) + reg (36 ch); one thread per (m, out)
// ---------------------------------------------------------------------------
__global__ __launch_bounds__(256) void proj_kernel(const u16* __restrict__ hid,
                                                   const float* __restrict__ conf_w,
                                                   const float* __restrict__ conf_b,
                                                   const float* __restrict__ reg_w,
                                                   const float* __restrict__ reg_b,
                                                   float* __restrict__ conf_out,
                                                   float* __restrict__ offs_out) {
    int idx = blockIdx.x * 256 + threadIdx.x;
    if (idx >= MDIM * 45) return;
    int m = idx / 45, o = idx - m * 45;
    const u16* hp = hid + (size_t)m * HID;
    const float* wp;
    float acc;
    if (o < NA) { wp = conf_w + o * HID;       acc = conf_b[o]; }
    else        { wp = reg_w + (o - NA) * HID; acc = reg_b[o - NA]; }
    #pragma unroll 8
    for (int h = 0; h < HID; ++h) acc += bf2f(hp[h]) * wp[h];
    int b = m >> 10, cell = m & 1023;
    if (o < NA) {
        conf_out[b * NANCH + cell * NA + o] = acc;
    } else {
        int ro = o - NA, a = ro >> 2, d = ro & 3;
        offs_out[(size_t)(b * NANCH + cell * NA + a) * 4 + d] = acc;
    }
}

// ---------------------------------------------------------------------------
// Anchors / IoU / loss
// ---------------------------------------------------------------------------
__device__ __forceinline__ void anchor_box(int n, float& x1, float& y1,
                                           float& x2, float& y2) {
    int a = n % NA, cell = n / NA;
    int y = cell >> 5, x = cell & 31;
    float aw = 2.0f * (float)(a / 3 + 1);          // scales 2,4,6
    float ah = aw * 0.5f * (float)(a % 3 + 1);     // * ratios 0.5,1,1.5
    float xc = (float)x + 0.5f, yc = (float)y + 0.5f;
    x1 = fminf(fmaxf(xc - aw * 0.5f, 0.0f), (float)WW);
    y1 = fminf(fmaxf(yc - ah * 0.5f, 0.0f), (float)HH);
    x2 = fminf(fmaxf(xc + aw * 0.5f, 0.0f), (float)WW);
    y2 = fminf(fmaxf(yc + ah * 0.5f, 0.0f), (float)HH);
}

__device__ __forceinline__ float iou_fn(float ax1, float ay1, float ax2, float ay2,
                                        float gx1, float gy1, float gx2, float gy2) {
    float iw = fmaxf(fminf(ax2, gx2) - fmaxf(ax1, gx1), 0.0f);
    float ih = fmaxf(fminf(ay2, gy2) - fmaxf(ay1, gy1), 0.0f);
    float inter = iw * ih;
    float aa = (ax2 - ax1) * (ay2 - ay1);
    float ag = (gx2 - gx1) * (gy2 - gy1);
    return inter / (aa + ag - inter + EPSF);
}

__device__ __forceinline__ float softplus_f(float x) {
    return fmaxf(x, 0.0f) + log1pf(expf(-fabsf(x)));
}

__global__ __launch_bounds__(256) void init_kernel(float* __restrict__ out_total,
                                                   float* __restrict__ maxpg,
                                                   u16* __restrict__ zpad) {
    int t = threadIdx.x;
    if (blockIdx.x == 0) {
        if (t == 0) out_total[0] = 0.0f;
        if (t < BATCH * NG) maxpg[t] = 0.0f;
        if (t < 32) zpad[t] = (u16)0;   // 64B zero pad for OOB im2col rows
    }
}

__global__ __launch_bounds__(256) void ioumax_kernel(const float* __restrict__ gtb,
                                                     float* __restrict__ maxpg) {
    int idx = blockIdx.x * 256 + threadIdx.x;
    if (idx >= BATCH * NANCH) return;
    int b = idx / NANCH, n = idx - b * NANCH;
    float ax1, ay1, ax2, ay2;
    anchor_box(n, ax1, ay1, ax2, ay2);
    for (int g = 0; g < NG; ++g) {
        const float* gp = gtb + ((size_t)b * NG + g) * 4;
        float iou = iou_fn(ax1, ay1, ax2, ay2,
                           gp[0] / SCALE_F, gp[1] / SCALE_F,
                           gp[2] / SCALE_F, gp[3] / SCALE_F);
        atomicMax((int*)&maxpg[b * NG + g], __float_as_int(iou));  // iou >= 0
    }
}

__global__ __launch_bounds__(256) void head_kernel(const float* __restrict__ gtb,
                                                   const float* __restrict__ maxpg,
                                                   const float* __restrict__ conf,
                                                   const float* __restrict__ offs,
                                                   float* __restrict__ out_total,
                                                   float* __restrict__ proposals,
                                                   float* __restrict__ pos_out) {
    __shared__ float red[256];
    int idx = blockIdx.x * 256 + threadIdx.x;
    float local = 0.0f;
    if (idx < BATCH * NANCH) {
        int b = idx / NANCH, n = idx - b * NANCH;
        float ax1, ay1, ax2, ay2;
        anchor_box(n, ax1, ay1, ax2, ay2);
        float acx = 0.5f * (ax1 + ax2), acy = 0.5f * (ay1 + ay2);
        float aw = ax2 - ax1, ah = ay2 - ay1;

        float best = -1.0f;
        int gi = 0;
        bool pos = false;
        for (int g = 0; g < NG; ++g) {
            const float* gp = gtb + ((size_t)b * NG + g) * 4;
            float iou = iou_fn(ax1, ay1, ax2, ay2,
                               gp[0] / SCALE_F, gp[1] / SCALE_F,
                               gp[2] / SCALE_F, gp[3] / SCALE_F);
            float mg = maxpg[b * NG + g];
            if (iou > POS_T) pos = true;
            if (iou == mg && mg > EPSF) pos = true;
            if (iou > best) { best = iou; gi = g; }
        }
        bool neg = (best < NEG_T) && !pos;

        float c = conf[idx];
        float cls = pos ? softplus_f(-c) : (neg ? softplus_f(c) : 0.0f);

        const float* op = offs + (size_t)idx * 4;
        float o0 = op[0], o1 = op[1], o2 = op[2], o3 = op[3];

        float reg = 0.0f;
        if (pos) {
            const float* gp = gtb + ((size_t)b * NG + gi) * 4;
            float g0 = gp[0] / SCALE_F, g1 = gp[1] / SCALE_F;
            float g2 = gp[2] / SCALE_F, g3 = gp[3] / SCALE_F;
            float gcx = 0.5f * (g0 + g2), gcy = 0.5f * (g1 + g3);
            float gw = g2 - g0, gh = g3 - g1;
            float t0 = (gcx - acx) / (aw + EPSF);
            float t1 = (gcy - acy) / (ah + EPSF);
            float t2 = logf((gw + EPSF) / (aw + EPSF));
            float t3 = logf((gh + EPSF) / (ah + EPSF));
            float d[4] = {o0 - t0, o1 - t1, o2 - t2, o3 - t3};
            #pragma unroll
            for (int k = 0; k < 4; ++k) {
                float ad = fabsf(d[k]);
                reg += (ad < 1.0f) ? 0.5f * d[k] * d[k] : ad - 0.5f;
            }
        }
        local = (W_CONF * cls + W_REG * reg) / (float)BATCH;

        // proposals (masked by pos)
        float pf  = pos ? 1.0f : 0.0f;
        float pcx = acx + o0 * aw, pcy = acy + o1 * ah;
        float pw  = aw * expf(o2), ph = ah * expf(o3);
        float* pp = proposals + (size_t)idx * 4;
        pp[0] = (pcx - 0.5f * pw) * pf;
        pp[1] = (pcy - 0.5f * ph) * pf;
        pp[2] = (pcx + 0.5f * pw) * pf;
        pp[3] = (pcy + 0.5f * ph) * pf;
        pos_out[idx] = pf;
    }
    // block reduction + single atomic per block
    red[threadIdx.x] = local;
    __syncthreads();
    for (int s = 128; s > 0; s >>= 1) {
        if (threadIdx.x < s) red[threadIdx.x] += red[threadIdx.x + s];
        __syncthreads();
    }
    if (threadIdx.x == 0) atomicAdd(out_total, red[0]);
}

// ---------------------------------------------------------------------------
// Launcher
// ---------------------------------------------------------------------------
extern "C" void kernel_launch(void* const* d_in, const int* in_sizes, int n_in,
                              void* d_out, int out_size, void* d_ws, size_t ws_size,
                              hipStream_t stream) {
    const float* fm     = (const float*)d_in[0];
    const float* gtb    = (const float*)d_in[1];
    // d_in[2] = gt_classes (unused by the reference loss)
    const float* conv1w = (const float*)d_in[3];
    const float* conv1b = (const float*)d_in[4];
    const float* confw  = (const float*)d_in[5];
    const float* confb  = (const float*)d_in[6];
    const float* regw   = (const float*)d_in[7];
    const float* regb   = (const float*)d_in[8];

    // workspace layout (bytes, 256-aligned blocks)
    char* ws = (char*)d_ws;
    size_t off = 0;
    u16* Wpack = (u16*)(ws + off); off += (size_t)HID * KFULL * sizeof(u16);     // 18.87 MB
    off = (off + 255) & ~(size_t)255;
    u16* Apack = (u16*)(ws + off); off += (size_t)MDIM * CIN * sizeof(u16);      // 16.78 MB
    off = (off + 255) & ~(size_t)255;
    u16* hid   = (u16*)(ws + off); off += (size_t)MDIM * HID * sizeof(u16);      // 4.19 MB
    off = (off + 255) & ~(size_t)255;
    float* confbuf = (float*)(ws + off); off += (size_t)BATCH * NANCH * sizeof(float);
    off = (off + 255) & ~(size_t)255;
    float* offsbuf = (float*)(ws + off); off += (size_t)BATCH * NANCH * 4 * sizeof(float);
    off = (off + 255) & ~(size_t)255;
    float* maxpg = (float*)(ws + off); off += (size_t)BATCH * NG * sizeof(float);
    off = (off + 255) & ~(size_t)255;
    u16* zpad = (u16*)(ws + off); off += 256;   // 64B zero pad (aligned)

    float* out_total = (float*)d_out;
    float* proposals = out_total + 1;
    float* pos_out   = proposals + (size_t)BATCH * NANCH * 4;

    init_kernel<<<1, 256, 0, stream>>>(out_total, maxpg, zpad);

    {   // pack conv1 weights
        int total = HID * KFULL;
        pack_w_kernel<<<(total + 255) / 256, 256, 0, stream>>>(conv1w, Wpack);
    }
    {   // pack feature map
        int total = BATCH * CIN * HH * WW;
        pack_fm_kernel<<<(total + 255) / 256, 256, 0, stream>>>(fm, Apack);
    }
    {   // WMMA implicit-GEMM conv + bias + ReLU (async LDS double-buffered)
        dim3 grid(MDIM / MT, HID / NT);   // 32 x 4
        conv_gemm_kernel<<<grid, 256, 0, stream>>>(Apack, Wpack, conv1b, zpad, hid);
    }
    {   // 1x1 heads
        int total = MDIM * 45;
        proj_kernel<<<(total + 255) / 256, 256, 0, stream>>>(hid, confw, confb,
                                                             regw, regb,
                                                             confbuf, offsbuf);
    }
    {   // per-GT max IoU
        int total = BATCH * NANCH;
        ioumax_kernel<<<(total + 255) / 256, 256, 0, stream>>>(gtb, maxpg);
    }
    {   // loss + proposals + pos
        int total = BATCH * NANCH;
        head_kernel<<<(total + 255) / 256, 256, 0, stream>>>(gtb, maxpg, confbuf,
                                                             offsbuf, out_total,
                                                             proposals, pos_out);
    }
}
